// MomNet_9706626089663
// MI455X (gfx1250) — compile-verified, matrix operand order checked by reading, not run on hardware
//
#include <hip/hip_runtime.h>

// MomNet forward for MI455X (gfx1250, wave32).
// All dense linears -> v_wmma_f32_16x16x32_f16. N=256 GEMMs compute a 16x32
// output per wave (A fragment reused for two wmmas). N and ACT are template
// params so B loads / C stores use immediate offsets. Gather-fused A operands
// (cond[2E,576], [ef|mom], clf input) avoid materializing concat matrices.
// All M are multiples of 16; all WMMA K-loops run in clean 32-wide blocks
// (first-layer inputs/weights zero-padded in workspace).
// Param flattening assumption: top-level dict insertion order; 'params' as a
// JAX pytree (sorted dict keys recursively, lists in order) -> 75 inputs.

typedef __attribute__((ext_vector_type(16))) _Float16 v16h;
typedef __attribute__((ext_vector_type(8)))  float    v8f;

namespace {

constexpr int kNodes = 50000;   // multiple of 16
constexpr int kE     = 150000;  // multiple of 16
constexpr int kE2    = 300000;  // multiple of 16

// ---------------- WMMA GEMM ----------------
// C[M,NT] = act(gatherA[M,K] @ W[K,NT] + bias)
// MODE 0: A row-major [M,K], K % 32 == 0
// MODE 1: A = [ef(64) | X1[s[row]](256) | X1[d[row]](256)]  (K = 576)
// MODE 2: A = [ef(64) | X2(mom, 1 col)]                     (K = 65)
// NT == 256 -> two N-tiles per wave (A fragment reused).
template <int MODE, int NT, int ACT>
__global__ void wmma_gemm(const float* __restrict__ A, const float* __restrict__ W,
                          const float* __restrict__ bias, float* __restrict__ C,
                          int M, int K,
                          const float* __restrict__ X1, const float* __restrict__ X2,
                          const int* __restrict__ s, const int* __restrict__ d) {
  constexpr int NW = (NT == 256) ? 2 : 1;       // N-tiles per wave
  constexpr int tilesN = NT / (16 * NW);
  int tiles = (M >> 4) * tilesN;
  int wid = blockIdx.x * (blockDim.x >> 5) + (threadIdx.x >> 5);
  if (wid >= tiles) return;                     // whole-wave uniform exit
  int tm = wid / tilesN, tn = wid % tilesN;
  int lane = threadIdx.x & 31;
  int half = lane >> 4;                         // ISA 16-bit fragment lane group
  int l16  = lane & 15;
  int row  = (tm << 4) + l16;                   // A row for this lane
  int nc0  = tn * (16 * NW) + l16;              // first B/C column for this lane
  int k0   = half << 3;                         // K run 0 start within block
  int k1   = 16 + (half << 3);                  // K run 1 start within block

  // hoisted per-lane region base pointers
  const float* pA;
  const float* pS = nullptr;
  const float* pD = nullptr;
  if (MODE == 0) {
    pA = A + (size_t)row * K;
  } else if (MODE == 1) {
    pA = A + (size_t)row * 64;
    pS = X1 + (size_t)s[row] * 256;
    pD = X1 + (size_t)d[row] * 256;
  } else {
    pA = A + (size_t)row * 64;
  }

  v8f acc0, acc1;
  {
    float b0 = bias ? bias[nc0] : 0.f;          // bias depends only on column
#pragma unroll
    for (int r = 0; r < 8; ++r) acc0[r] = b0;
    if (NW == 2) {
      float b1 = bias ? bias[nc0 + 16] : 0.f;
#pragma unroll
      for (int r = 0; r < 8; ++r) acc1[r] = b1;
    }
  }

  int kfull = K & ~31;
  for (int kk = 0; kk < kfull; kk += 32) {
    const float* pa;
    if (MODE == 1)   // region boundaries (64, 320) are 32-aligned: whole block
      pa = (kk < 64) ? (pA + kk) : ((kk < 320) ? (pS + (kk - 64)) : (pD + (kk - 320)));
    else
      pa = pA + kk;
    const float* pw = W + (size_t)kk * NT + nc0;
    v16h av, bv0, bv1;
#pragma unroll
    for (int i = 0; i < 8; ++i) {               // two contiguous 32B runs per lane
      av[i]      = (_Float16)pa[k0 + i];
      av[i + 8]  = (_Float16)pa[k1 + i];
      bv0[i]     = (_Float16)pw[(k0 + i) * NT]; // immediate strides (NT const)
      bv0[i + 8] = (_Float16)pw[(k1 + i) * NT];
      if (NW == 2) {
        bv1[i]     = (_Float16)pw[(k0 + i) * NT + 16];
        bv1[i + 8] = (_Float16)pw[(k1 + i) * NT + 16];
      }
    }
    acc0 = __builtin_amdgcn_wmma_f32_16x16x32_f16(false, av, false, bv0,
                                                  (short)0, acc0, false, false);
    if (NW == 2)
      acc1 = __builtin_amdgcn_wmma_f32_16x16x32_f16(false, av, false, bv1,
                                                    (short)0, acc1, false, false);
  }

  if (MODE == 2) {
    // rank-1 update for the single extra column (k = 64): A[row,64] = mom[row]
    float m8[8];
#pragma unroll
    for (int r = 0; r < 8; ++r) m8[r] = X2[(tm << 4) + r + (half << 3)];
    float wv0 = W[(size_t)64 * NT + nc0];
#pragma unroll
    for (int r = 0; r < 8; ++r) acc0[r] += m8[r] * wv0;
    if (NW == 2) {
      float wv1 = W[(size_t)64 * NT + nc0 + 16];
#pragma unroll
      for (int r = 0; r < 8; ++r) acc1[r] += m8[r] * wv1;
    }
  }

  // D fragment: VGPR r -> row r + 8*half; compile-time store strides
  float* pc = C + (size_t)((tm << 4) + (half << 3)) * NT + nc0;
#pragma unroll
  for (int r = 0; r < 8; ++r) {
    float v0 = acc0[r];
    if (ACT) v0 = fmaxf(v0, 0.f);
    pc[r * NT] = v0;
    if (NW == 2) {
      float v1 = acc1[r];
      if (ACT) v1 = fmaxf(v1, 0.f);
      pc[r * NT + 16] = v1;
    }
  }
}

// ---------------- padding / setup kernels ----------------
__global__ void k_pad_w32(const float* __restrict__ w, float* __restrict__ wp,
                          int Krows, int N) {
  int i = blockIdx.x * blockDim.x + threadIdx.x;   // over 32*N
  if (i >= 32 * N) return;
  int k = i / N, n = i - k * N;
  wp[i] = (k < Krows) ? w[(size_t)k * N + n] : 0.f;
}

__global__ void k_pad_rows32(const float* __restrict__ x, float* __restrict__ xp,
                             int M, int Kin) {
  long i = (long)blockIdx.x * blockDim.x + threadIdx.x;  // over M*32
  if (i >= (long)M * 32) return;
  int r = (int)(i >> 5), c = (int)(i & 31);
  xp[i] = (c < Kin) ? x[(size_t)r * Kin + c] : 0.f;
}

__global__ void k_fill_u32(unsigned* p, unsigned v, int n) {
  int i = blockIdx.x * blockDim.x + threadIdx.x;
  if (i < n) p[i] = v;
}

__global__ void k_build_edges(const int* __restrict__ ei, int* __restrict__ srcD,
                              int* __restrict__ dstD, int E) {
  int e = blockIdx.x * blockDim.x + threadIdx.x;
  if (e >= E) return;
  int a = ei[e], b = ei[E + e];
  srcD[e] = a;     dstD[e] = b;       // forward directed edge
  srcD[E + e] = b; dstD[E + e] = a;   // reversed duplicate
}

// efin is [2E, 32], zero-padded beyond col 7
__global__ void k_build_efin(const float* __restrict__ ea, const float* __restrict__ lg,
                             float* __restrict__ efin, int E) {
  int e = blockIdx.x * blockDim.x + threadIdx.x;
  if (e >= E) return;
  float v[32];
#pragma unroll
  for (int j = 0; j < 32; ++j) v[j] = 0.f;
#pragma unroll
  for (int j = 0; j < 7; ++j) v[j] = ea[(size_t)e * 7 + j];
  v[7] = lg[e];
#pragma unroll
  for (int j = 0; j < 32; ++j) {
    efin[(size_t)e * 32 + j] = v[j];
    efin[(size_t)(E + e) * 32 + j] = v[j];
  }
}

// ---------------- attention kernels ----------------
__device__ __forceinline__ unsigned f2ord(float f) {
  unsigned u = __float_as_uint(f);
  return (u & 0x80000000u) ? ~u : (u | 0x80000000u);
}
__device__ __forceinline__ float ord2f(unsigned m) {
  return __uint_as_float((m & 0x80000000u) ? (m ^ 0x80000000u) : ~m);
}

__global__ void k_alpha(const float* __restrict__ Q, const float* __restrict__ Kf,
                        const float* __restrict__ Ep, const int* __restrict__ srcD,
                        const int* __restrict__ dstD, float* __restrict__ alpha,
                        unsigned* __restrict__ amaxU, int E2) {
  int idx = blockIdx.x * blockDim.x + threadIdx.x;
  if (idx >= E2 * 4) return;
  int e = idx >> 2, h = idx & 3;
  int s = srcD[e], d = dstD[e];
  const float4* q4 = (const float4*)(Q  + (size_t)d * 256 + h * 64);
  const float4* k4 = (const float4*)(Kf + (size_t)s * 256 + h * 64);
  const float4* e4 = (const float4*)(Ep + (size_t)e * 256 + h * 64);
  float acc = 0.f;
#pragma unroll 4
  for (int c = 0; c < 16; ++c) {
    float4 qa = q4[c], kb = k4[c], eb = e4[c];
    acc += qa.x * (kb.x + eb.x) + qa.y * (kb.y + eb.y) +
           qa.z * (kb.z + eb.z) + qa.w * (kb.w + eb.w);
  }
  float a = acc * 0.125f;               // / sqrt(64)
  alpha[idx] = a;
  atomicMax(&amaxU[(size_t)d * 4 + h], f2ord(a));
}

__global__ void k_expsum(float* __restrict__ alpha, const unsigned* __restrict__ amaxU,
                         const int* __restrict__ dstD, float* __restrict__ denom, int E2) {
  int idx = blockIdx.x * blockDim.x + threadIdx.x;
  if (idx >= E2 * 4) return;
  int e = idx >> 2, h = idx & 3;
  int d = dstD[e];
  float m = ord2f(amaxU[(size_t)d * 4 + h]);
  float ex = expf(alpha[idx] - m);
  alpha[idx] = ex;
  atomicAdd(&denom[(size_t)d * 4 + h], ex);
}

__global__ void k_msg(const float* __restrict__ V, const float* __restrict__ Ep,
                      const float* __restrict__ alpha, const float* __restrict__ denom,
                      const int* __restrict__ srcD, const int* __restrict__ dstD,
                      float* __restrict__ out, int E2) {
  long idx = (long)blockIdx.x * blockDim.x + threadIdx.x;   // over E2*64 quads
  if (idx >= (long)E2 * 64) return;
  int e = (int)(idx >> 6);
  int jq = (int)(idx & 63);            // quad index within row (4 floats each)
  int h = jq >> 4;
  int s = srcD[e], d = dstD[e];
  float w = alpha[(size_t)e * 4 + h] / (denom[(size_t)d * 4 + h] + 1e-16f);
  const float4 vv = ((const float4*)(V  + (size_t)s * 256))[jq];
  const float4 ee = ((const float4*)(Ep + (size_t)e * 256))[jq];
  float* o = out + (size_t)d * 256 + jq * 4;
  atomicAdd(o + 0, (vv.x + ee.x) * w);
  atomicAdd(o + 1, (vv.y + ee.y) * w);
  atomicAdd(o + 2, (vv.z + ee.z) * w);
  atomicAdd(o + 3, (vv.w + ee.w) * w);
}

// ---------------- layernorm / small linears ----------------
__global__ void k_layernorm(float* __restrict__ x, const float* __restrict__ g,
                            const float* __restrict__ b, int M, int D, int relu) {
  int row = blockIdx.x * (blockDim.x >> 5) + (threadIdx.x >> 5);  // wave per row
  if (row >= M) return;
  int lane = threadIdx.x & 31;
  float s = 0.f, s2 = 0.f;
  for (int j = lane; j < D; j += 32) {
    float v = x[(size_t)row * D + j];
    s += v; s2 += v * v;
  }
  for (int o = 16; o > 0; o >>= 1) {
    s  += __shfl_xor(s,  o, 32);
    s2 += __shfl_xor(s2, o, 32);
  }
  float mean = s / D;
  float var = fmaxf(s2 / D - mean * mean, 0.f);
  float inv = rsqrtf(var + 1e-5f);
  for (int j = lane; j < D; j += 32) {
    float v = (x[(size_t)row * D + j] - mean) * inv * g[j] + b[j];
    if (relu) v = fmaxf(v, 0.f);
    x[(size_t)row * D + j] = v;
  }
}

__global__ void k_small_linear(const float* __restrict__ A, const float* __restrict__ W,
                               const float* __restrict__ bias, float* __restrict__ C,
                               int M, int K, int N, int act, int mode,
                               const float* __restrict__ X1,
                               const int* __restrict__ i0, const int* __restrict__ i1) {
  long idx = (long)blockIdx.x * blockDim.x + threadIdx.x;
  if (idx >= (long)M * N) return;
  int row = (int)(idx / N), n = (int)(idx % N);
  float acc = bias ? bias[n] : 0.f;
  if (mode == 0) {
    const float* pa = A + (size_t)row * K;
    for (int k = 0; k < K; ++k) acc += pa[k] * W[(size_t)k * N + n];
  } else {  // clf gather: [nf[i0[row]] | nf[i1[row]]]
    const float* p0 = X1 + (size_t)i0[row] * 64;
    const float* p1 = X1 + (size_t)i1[row] * 64;
    for (int k = 0; k < 64; ++k) acc += p0[k] * W[(size_t)k * N + n];
    for (int k = 0; k < 64; ++k) acc += p1[k] * W[(size_t)(64 + k) * N + n];
  }
  if (act) acc = fmaxf(acc, 0.f);
  C[(size_t)row * N + n] = acc;
}

__global__ void k_mom_avg(float* __restrict__ mom, float* __restrict__ out_avg, int E) {
  int e = blockIdx.x * blockDim.x + threadIdx.x;
  if (e >= E) return;
  float a = 0.5f * (mom[e] + mom[E + e]);
  mom[e] = a; mom[E + e] = a;
  out_avg[e] = a;
}

// ---------------- host helpers ----------------
inline int ceil_div(long a, long b) { return (int)((a + b - 1) / b); }

// MODE 0 dispatchers
void g0_n64(const float* A, const float* W, const float* b, float* C,
            int M, int K, int act, hipStream_t st) {
  int tiles = (M / 16) * (64 / 16);
  if (act)
    wmma_gemm<0, 64, 1><<<ceil_div(tiles, 8), 256, 0, st>>>(A, W, b, C, M, K,
                                                            nullptr, nullptr, nullptr, nullptr);
  else
    wmma_gemm<0, 64, 0><<<ceil_div(tiles, 8), 256, 0, st>>>(A, W, b, C, M, K,
                                                            nullptr, nullptr, nullptr, nullptr);
}
void g0_n256(const float* A, const float* W, const float* b, float* C,
             int M, int K, hipStream_t st) {
  int tiles = (M / 16) * (256 / 32);    // 2 N-tiles per wave
  wmma_gemm<0, 256, 0><<<ceil_div(tiles, 8), 256, 0, st>>>(A, W, b, C, M, K,
                                                           nullptr, nullptr, nullptr, nullptr);
}
void g1_n64(const float* ef, const float* W, const float* b, float* C,
            int M, const float* nfbig, const int* s, const int* d, hipStream_t st) {
  int tiles = (M / 16) * (64 / 16);
  wmma_gemm<1, 64, 0><<<ceil_div(tiles, 8), 256, 0, st>>>(ef, W, b, C, M, 576,
                                                          nfbig, nullptr, s, d);
}
void g2_n256(const float* ef, const float* W, float* C,
             int M, const float* mom, hipStream_t st) {
  int tiles = (M / 16) * (256 / 32);
  wmma_gemm<2, 256, 0><<<ceil_div(tiles, 8), 256, 0, st>>>(ef, W, nullptr, C, M, 65,
                                                           nullptr, mom, nullptr, nullptr);
}

void layernorm(float* x, const float* g, const float* b, int M, int D, int relu,
               hipStream_t st) {
  k_layernorm<<<ceil_div(M, 8), 256, 0, st>>>(x, g, b, M, D, relu);
}

void tconv(const float* nfin,
           const float* Wq, const float* bq, const float* Wk, const float* bk,
           const float* Wv, const float* bv, const float* Ws, const float* bs,
           const float* We, int emode, const float* momp,
           const float* ef, const int* srcD, const int* dstD,
           float* Q, float* Kf, float* Vf, float* attn, float* Ep,
           float* alpha, unsigned* amaxU, float* denom, hipStream_t st) {
  g0_n256(nfin, Wq, bq, Q,    kNodes, 64, st);
  g0_n256(nfin, Wk, bk, Kf,   kNodes, 64, st);
  g0_n256(nfin, Wv, bv, Vf,   kNodes, 64, st);
  g0_n256(nfin, Ws, bs, attn, kNodes, 64, st);   // skip pre-fills output
  if (emode == 0) g0_n256(ef, We, nullptr, Ep, kE2, 64, st);
  else            g2_n256(ef, We, Ep, kE2, momp, st);
  k_fill_u32<<<ceil_div(kNodes * 4, 256), 256, 0, st>>>(amaxU, 0u, kNodes * 4);
  k_fill_u32<<<ceil_div(kNodes * 4, 256), 256, 0, st>>>((unsigned*)denom, 0u, kNodes * 4);
  k_alpha<<<ceil_div((long)kE2 * 4, 256), 256, 0, st>>>(Q, Kf, Ep, srcD, dstD, alpha, amaxU, kE2);
  k_expsum<<<ceil_div((long)kE2 * 4, 256), 256, 0, st>>>(alpha, amaxU, dstD, denom, kE2);
  k_msg<<<ceil_div((long)kE2 * 64, 256), 256, 0, st>>>(Vf, Ep, alpha, denom, srcD, dstD, attn, kE2);
}

} // namespace

extern "C" void kernel_launch(void* const* d_in, const int* in_sizes, int n_in,
                              void* d_out, int out_size, void* d_ws, size_t ws_size,
                              hipStream_t stream) {
  (void)in_sizes; (void)n_in; (void)out_size; (void)ws_size;
  auto F = [&](int i) { return (const float*)d_in[i]; };
  const float* x    = F(0);
  const float* eatt = F(1);
  const float* elog = F(2);
  const int*   ei   = (const int*)d_in[3];      // [2, E]

  // params pytree (sorted keys): agg, clf, edge_emb, fc_mom, layers, node_emb,
  //                              norm_combined, proj_node
  const float* agg_e_w = F(4);
  const float* agg_k_b = F(5);  const float* agg_k_w = F(6);
  const float* agg_q_b = F(7);  const float* agg_q_w = F(8);
  const float* agg_s_b = F(9);  const float* agg_s_w = F(10);
  const float* agg_v_b = F(11); const float* agg_v_w = F(12);
  const float* clf0_b = F(13);  const float* clf0_w = F(14);
  const float* clf1_b = F(15);  const float* clf1_w = F(16);
  const float* ee0_b = F(17);   const float* ee0_w = F(18);
  const float* ee1_b = F(19);   const float* ee1_w = F(20);
  const float* fm0_b = F(21);   const float* fm0_w = F(22);
  const float* fm1_b = F(23);   const float* fm1_w = F(24);
  const float* fmln0_b = F(25); const float* fmln0_g = F(26);
  const float* fmln1_b = F(27); const float* fmln1_g = F(28);
  struct Lp {
    const float *ce_w, *ck_b, *ck_w, *cq_b, *cq_w, *cs_b, *cs_w, *cv_b, *cv_w;
    const float *et0_b, *et0_w, *et1_b, *et1_w, *etln0_b, *etln0_g, *etln1_b, *etln1_g;
    const float *p_b, *p_w;
  } L[2];
  for (int l = 0; l < 2; ++l) {
    int o = 29 + 19 * l;
    L[l] = { F(o+0), F(o+1), F(o+2), F(o+3), F(o+4), F(o+5), F(o+6), F(o+7), F(o+8),
             F(o+9), F(o+10), F(o+11), F(o+12), F(o+13), F(o+14), F(o+15), F(o+16),
             F(o+17), F(o+18) };
  }
  const float* ne0_b = F(67); const float* ne0_w = F(68);
  const float* ne1_b = F(69); const float* ne1_w = F(70);
  const float* nc_b  = F(71); const float* nc_g  = F(72);
  const float* pn_b  = F(73); const float* pn_w  = F(74);

  // ---- workspace carve-up ----
  float* ws = (float*)d_ws;
  size_t off = 0;
  auto take = [&](size_t n) { float* p = ws + off; off += n; return p; };
  float* nf    = take((size_t)kNodes * 64);
  float* ef    = take((size_t)kE2 * 64);
  float* ef2   = take((size_t)kE2 * 64);
  float* Q     = take((size_t)kNodes * 256);
  float* Kf    = take((size_t)kNodes * 256);
  float* Vf    = take((size_t)kNodes * 256);
  float* attn  = take((size_t)kNodes * 256);   // = nf_big
  float* Ep    = take((size_t)kE2 * 256);
  float* alpha = take((size_t)kE2 * 4);
  unsigned* amaxU = (unsigned*)take((size_t)kNodes * 4);
  float* denom = take((size_t)kNodes * 4);
  int* srcD = (int*)take((size_t)kE2);
  int* dstD = (int*)take((size_t)kE2);
  float* efin = take((size_t)kE2 * 32);        // padded edge-emb input; later fc_mom hidden
  float* xpad = take((size_t)kNodes * 32);     // x zero-padded 16->32
  float* wp0  = take((size_t)32 * 64);         // ne0_w padded 16->32 rows
  float* wp1  = take((size_t)32 * 64);         // ee0_w padded 8->32 rows
  float* mom  = take((size_t)kE2);
  float* clfh = take((size_t)kE * 16);

  float* out_avg  = (float*)d_out;             // [E]
  float* out_edge = (float*)d_out + kE;        // [E]

  k_build_edges<<<ceil_div(kE, 256), 256, 0, stream>>>(ei, srcD, dstD, kE);
  k_build_efin<<<ceil_div(kE, 256), 256, 0, stream>>>(eatt, elog, efin, kE);
  k_pad_rows32<<<ceil_div((long)kNodes * 32, 256), 256, 0, stream>>>(x, xpad, kNodes, 16);
  k_pad_w32<<<ceil_div(32 * 64, 256), 256, 0, stream>>>(ne0_w, wp0, 16, 64);
  k_pad_w32<<<ceil_div(32 * 64, 256), 256, 0, stream>>>(ee0_w, wp1, 8, 64);

  // node_emb (Q buffer doubles as [N,64] hidden)
  g0_n64(xpad, wp0, ne0_b, Q,  kNodes, 32, 1, stream);
  g0_n64(Q, ne1_w, ne1_b, nf,  kNodes, 64, 1, stream);
  // edge_emb
  g0_n64(efin, wp1, ee0_b, ef2, kE2, 32, 1, stream);
  g0_n64(ef2,  ee1_w, ee1_b, ef, kE2, 64, 1, stream);

  for (int l = 0; l < 2; ++l) {
    tconv(nf, L[l].cq_w, L[l].cq_b, L[l].ck_w, L[l].ck_b, L[l].cv_w, L[l].cv_b,
          L[l].cs_w, L[l].cs_b, L[l].ce_w, 0, nullptr,
          ef, srcD, dstD, Q, Kf, Vf, attn, Ep, alpha, amaxU, denom, stream);
    // edge_trans: fused cond-gather GEMM (K=576), LN+ReLU, 64x64 GEMM, LN+ReLU
    g1_n64(ef, L[l].et0_w, L[l].et0_b, ef2, kE2, attn, srcD, dstD, stream);
    layernorm(ef2, L[l].etln0_g, L[l].etln0_b, kE2, 64, 1, stream);
    g0_n64(ef2, L[l].et1_w, L[l].et1_b, ef, kE2, 64, 0, stream);
    layernorm(ef, L[l].etln1_g, L[l].etln1_b, kE2, 64, 1, stream);
    // proj: nf = attn @ Wp + bp
    g0_n64(attn, L[l].p_w, L[l].p_b, nf, kNodes, 256, 0, stream);
  }

  // fc_mom: 64->8 (LN,ReLU), 8->1 (LN).  efin reused as [2E,8] hidden.
  k_small_linear<<<ceil_div((long)kE2 * 8, 256), 256, 0, stream>>>(
      ef, fm0_w, fm0_b, efin, kE2, 64, 8, 0, 0, nullptr, nullptr, nullptr);
  layernorm(efin, fmln0_g, fmln0_b, kE2, 8, 1, stream);
  k_small_linear<<<ceil_div((long)kE2, 256), 256, 0, stream>>>(
      efin, fm1_w, fm1_b, mom, kE2, 8, 1, 0, 0, nullptr, nullptr, nullptr);
  layernorm(mom, fmln1_g, fmln1_b, kE2, 1, 0, stream);
  k_mom_avg<<<ceil_div(kE, 256), 256, 0, stream>>>(mom, out_avg, kE);

  // agg tconv: edge features [ef | mom] (K=65: 2 WMMA blocks + rank-1)
  tconv(nf, agg_q_w, agg_q_b, agg_k_w, agg_k_b, agg_v_w, agg_v_b,
        agg_s_w, agg_s_b, agg_e_w, 2, mom,
        ef, srcD, dstD, Q, Kf, Vf, attn, Ep, alpha, amaxU, denom, stream);
  layernorm(attn, nc_g, nc_b, kNodes, 256, 0, stream);
  g0_n64(attn, pn_w, pn_b, nf, kNodes, 256, 0, stream);

  // clf: gather [nf[i0] | nf[i1]] -> 16 (ReLU) -> 1
  k_small_linear<<<ceil_div((long)kE * 16, 256), 256, 0, stream>>>(
      nullptr, clf0_w, clf0_b, clfh, kE, 128, 16, 1, 1, nf, ei, ei + kE);
  k_small_linear<<<ceil_div((long)kE, 256), 256, 0, stream>>>(
      clfh, clf1_w, clf1_b, out_edge, kE, 16, 1, 0, 0, nullptr, nullptr, nullptr);
}
